// DeepKalman_712964571464
// MI455X (gfx1250) — compile-verified, matrix operand order checked by reading, not run on hardware
//
#include <hip/hip_runtime.h>

// ---------------------------------------------------------------------------
// Deep Kalman Filter ELBO for MI455X (gfx1250, wave32, WMMA, TDM).
// B=256, T=512, S=64, O=128, H=128, 4H=512.
// bf16 WMMA (v_wmma_f32_16x16x32_bf16) with fp32 accumulation; TDM staging
// of LSTM gate tiles; async-to-LDS staging of GEMM input tiles.
// ---------------------------------------------------------------------------

#define B_   256
#define T_   512
#define S_   64
#define O_   128
#define H_   128
#define G4H  512
#define M_   (B_ * T_)           // 131072 flattened (b,t) rows
#define LOG2PI_ 1.8378770664093453f

typedef __attribute__((ext_vector_type(16))) __bf16 v16bf;
typedef __attribute__((ext_vector_type(8)))  float  v8f;
typedef __attribute__((ext_vector_type(4)))  unsigned int v4u;
typedef __attribute__((ext_vector_type(8)))  int    v8i;
typedef __attribute__((ext_vector_type(4)))  int    v4i;
typedef int v4i_vs __attribute__((vector_size(16)));   // gcc-style, matches builtin proto

#if __has_builtin(__builtin_amdgcn_global_load_async_to_lds_b128)
#define HAS_ASYNC_LDS 1
#else
#define HAS_ASYNC_LDS 0
#endif
#if __has_builtin(__builtin_amdgcn_tensor_load_to_lds)
#define HAS_TDM 1
#else
#define HAS_TDM 0
#endif

// wave id as a compiler-provable scalar (forces s_cbranch, not exec masking)
static __device__ __forceinline__ int wave_id() {
  return __builtin_amdgcn_readfirstlane((int)(threadIdx.x >> 5));
}

// ---------------------------------------------------------------------------
// Async / TDM staging helpers
// ---------------------------------------------------------------------------
static __device__ __forceinline__ void cp16_async(void* lds, const void* g) {
#if HAS_ASYNC_LDS
  __builtin_amdgcn_global_load_async_to_lds_b128(
      (__attribute__((address_space(1))) v4i_vs*)g,
      (__attribute__((address_space(3))) v4i_vs*)lds, 0, 0);
#else
  *(float4*)lds = *(const float4*)g;
#endif
}

static __device__ __forceinline__ void wait_async0() {
#if HAS_ASYNC_LDS
#if __has_builtin(__builtin_amdgcn_s_wait_asynccnt)
  __builtin_amdgcn_s_wait_asynccnt(0);
#else
  asm volatile("s_wait_asynccnt 0x0" ::: "memory");
#endif
#endif
}

// Stage one contiguous 16x512 f32 gate tile (32KB) into LDS with a 2-dword
// pad every 256 dwords (LDS row stride 516 dwords -> bank-conflict free).
// TDM path: single tensor_load_to_lds issued by wave 0 (D# per ISA ch.8).
static __device__ __forceinline__ void stage_xk_tile(const float* __restrict__ gsrc,
                                                     float* __restrict__ ldsbuf) {
#if HAS_TDM
  if (wave_id() == 0) {
    unsigned long long ga = (unsigned long long)(size_t)gsrc;
    unsigned laddr = (unsigned)(size_t)(__attribute__((address_space(3))) char*)ldsbuf;
    v4u g0;
    g0[0] = 1u;                                           // count=1, user desc
    g0[1] = laddr;                                        // lds_addr
    g0[2] = (unsigned)(ga & 0xffffffffu);                 // global_addr[31:0]
    g0[3] = (unsigned)((ga >> 32) & 0x01ffffffu)          // global_addr[56:32]
          | 0x80000000u;                                  // type=2 (image)
    v8i g1;
    g1[0] = (int)0x03D20000u;  // data_size=4B | pad_en | pad_interval=256dw | pad=2dw
    g1[1] = (int)0x20000000u;  // tensor_dim0 = 8192 (bits 79:48, low half here)
    g1[2] = (int)0x00010000u;  // tensor_dim0 hi = 0, tensor_dim1 = 1 (low half)
    g1[3] = (int)0x20000000u;  // tensor_dim1 hi = 0, tile_dim0 = 8192
    g1[4] = 1;                 // tile_dim1 = 1, tile_dim2 = 0
    g1[5] = 8192;              // tensor_dim0_stride (low 32)
    g1[6] = 0;
    g1[7] = 0;
    v4i g2 = {0, 0, 0, 0};
    v4i g3 = {0, 0, 0, 0};
    v8i g4 = {0, 0, 0, 0, 0, 0, 0, 0};
    __builtin_amdgcn_tensor_load_to_lds(g0, g1, g2, g3, g4, 0);
  }
#else
  for (int ch = threadIdx.x; ch < 512; ch += blockDim.x) {   // 512 x 16B chunks
    const int d  = ch * 4;                 // flat dword index
    const int dd = d + ((d >> 8) << 1);    // pad-adjusted LDS dword index
    cp16_async(&ldsbuf[dd], gsrc + d);
  }
#endif
}

static __device__ __forceinline__ void stage_wait() {
#if HAS_TDM
  __builtin_amdgcn_s_wait_tensorcnt(0);
#else
  wait_async0();
#endif
}

// padded LDS index for the gate tile: row-major (16 x 512) with pad
static __device__ __forceinline__ int xidx(int row, int col) {
  return row * 516 + col + ((col >> 8) << 1);
}

// ---------------------------------------------------------------------------
// WMMA fragment helpers (layouts per CDNA5 ISA 7.12.2, wave32).
// ---------------------------------------------------------------------------
static __device__ __forceinline__ v16bf frag_A_f32(const float* src, int ld, int k0) {
  const int lane = threadIdx.x & 31;
  const int m  = lane & 15;
  const int kb = (lane < 16) ? 0 : 8;
  v16bf a;
#pragma unroll
  for (int e = 0; e < 8; ++e) a[e]     = (__bf16)src[m * ld + k0 + kb + e];
#pragma unroll
  for (int e = 0; e < 8; ++e) a[8 + e] = (__bf16)src[m * ld + k0 + 16 + kb + e];
  return a;
}

static __device__ __forceinline__ v16bf frag_A_bf16(const __bf16* src, int ld, int k0) {
  const int lane = threadIdx.x & 31;
  const int m  = lane & 15;
  const int kb = (lane < 16) ? 0 : 8;
  v16bf a;
#pragma unroll
  for (int e = 0; e < 8; ++e) a[e]     = src[m * ld + k0 + kb + e];
#pragma unroll
  for (int e = 0; e < 8; ++e) a[8 + e] = src[m * ld + k0 + 16 + kb + e];
  return a;
}

static __device__ __forceinline__ v16bf frag_B_bf16(const __bf16* src, int ld, int k0, int n0) {
  const int lane = threadIdx.x & 31;
  const int n  = n0 + (lane & 15);
  const int kb = (lane < 16) ? 0 : 8;
  v16bf b;
#pragma unroll
  for (int e = 0; e < 8; ++e) b[e]     = src[(k0 + kb + e) * ld + n];
#pragma unroll
  for (int e = 0; e < 8; ++e) b[8 + e] = src[(k0 + 16 + kb + e) * ld + n];
  return b;
}

static __device__ __forceinline__ v8f wmma_bf16(v16bf a, v16bf b, v8f c) {
  return __builtin_amdgcn_wmma_f32_16x16x32_bf16(false, a, false, b, (short)0, c,
                                                 false, false);
}

// ---------------------------------------------------------------------------
// Branchless math on raw HW transcendentals (v_exp_f32 / v_log_f32 / v_rcp_f32)
// ---------------------------------------------------------------------------
static __device__ __forceinline__ float fexp_(float x) {
  return __builtin_amdgcn_exp2f(x * 1.4426950408889634f);
}
static __device__ __forceinline__ float flog_(float x) {
  return 0.6931471805599453f * __builtin_amdgcn_logf(x);
}
static __device__ __forceinline__ float softplus_(float x) {
  return fmaxf(x, 0.0f) + flog_(1.0f + fexp_(-fabsf(x)));
}
static __device__ __forceinline__ float sigmoid_(float x) {
  return __builtin_amdgcn_rcpf(1.0f + fexp_(-x));
}
static __device__ __forceinline__ float tanh_(float x) {
  const float e = fexp_(-2.0f * fabsf(x));
  const float r = (1.0f - e) * __builtin_amdgcn_rcpf(1.0f + e);
  return copysignf(r, x);
}

// ---------------------------------------------------------------------------
// K0: fp32 -> bf16 weight conversion
// ---------------------------------------------------------------------------
__global__ void cvt_kernel(const float* __restrict__ src, __bf16* __restrict__ dst, int n) {
  int i = blockIdx.x * blockDim.x + threadIdx.x;
  if (i < n) dst[i] = (__bf16)src[i];
}

// ---------------------------------------------------------------------------
// K1: prior rollout (sequential scan over T, parallel over batch tiles).
// 16 blocks x 256 threads (8 waves): wave = {mean/scale} x {4 N-tiles}.
// Weight B-fragments live in registers across all 512 steps.
// ---------------------------------------------------------------------------
__global__ void prior_kernel(const float* __restrict__ init_mean,
                             const float* __restrict__ bf_m,
                             const float* __restrict__ bf_s,
                             const __bf16* __restrict__ Wm,
                             const __bf16* __restrict__ Ws,
                             float* __restrict__ prior_mean,
                             float* __restrict__ prior_scale) {
  __shared__ float mcur[16][S_ + 4];
  const int b0   = blockIdx.x * 16;
  const int wave = wave_id();            // scalar: s_cbranch, no exec masking
  const int lane = threadIdx.x & 31;
  const int mat  = wave >> 2;            // 0 = mean net, 1 = scale net
  const int n0   = (wave & 3) * 16;
  const __bf16* W    = mat ? Ws : Wm;
  const float*  bias = mat ? bf_s : bf_m;
  float* gdst = mat ? prior_scale : prior_mean;   // uniform pointer select

  v16bf wb0 = frag_B_bf16(W, S_, 0, n0);
  v16bf wb1 = frag_B_bf16(W, S_, 32, n0);

  for (int i = threadIdx.x; i < 16 * S_; i += blockDim.x)
    mcur[i >> 6][i & 63] = init_mean[(size_t)(b0 + (i >> 6)) * S_ + (i & 63)];
  __syncthreads();

  const int   n  = lane & 15;
  const float bn = bias[n0 + n];

  for (int t = 0; t < T_; ++t) {
    v16bf a0 = frag_A_f32(&mcur[0][0], S_ + 4, 0);
    v16bf a1 = frag_A_f32(&mcur[0][0], S_ + 4, 32);
    v8f acc = {};
    acc = wmma_bf16(a0, wb0, acc);
    acc = wmma_bf16(a1, wb1, acc);
    __syncthreads();   // all waves finished reading mcur
#pragma unroll
    for (int v = 0; v < 8; ++v) {
      const int    m   = (lane < 16) ? v : v + 8;
      const float  val = acc[v] + bn;
      const size_t g   = ((size_t)(b0 + m) * T_ + t) * S_ + n0 + n;
      gdst[g] = mat ? softplus_(val) : val;
      if (mat == 0) mcur[m][n0 + n] = val;   // recurrence carries the mean
    }
    __syncthreads();
  }
}

// ---------------------------------------------------------------------------
// K2: xk = obs @ K_lstm + b_lstm, written in LSTM-tile layout:
//   xk[((bblk*T + t)*16 + brow)*512 + n]   (one contiguous 32KB tile per step)
// grid (8192, 8), 128 threads (4 waves). obs tile staged via async-to-LDS.
// ---------------------------------------------------------------------------
__global__ void xk_kernel(const float* __restrict__ obs,
                          const __bf16* __restrict__ K,
                          const float* __restrict__ b_lstm,
                          float* __restrict__ xk) {
  __shared__ float xt[16][O_ + 4];
  const int m0   = blockIdx.x * 16;        // 16 consecutive flat rows: same b
  const int wave = wave_id();
  const int lane = threadIdx.x & 31;
  const int n0   = blockIdx.y * 64 + wave * 16;

  // stage 16x128 f32 obs tile, row-wise 16B chunks (512 chunks)
  for (int ch = threadIdx.x; ch < 512; ch += blockDim.x) {
    const int r = ch >> 5, w = ch & 31;
    cp16_async(&xt[r][w * 4], obs + (size_t)(m0 + r) * O_ + w * 4);
  }
  wait_async0();
  __syncthreads();

  v8f acc = {};
#pragma unroll
  for (int kf = 0; kf < 4; ++kf) {
    v16bf a = frag_A_f32(&xt[0][0], O_ + 4, kf * 32);
    v16bf b = frag_B_bf16(K, G4H, kf * 32, n0);
    acc = wmma_bf16(a, b, acc);
  }
  const int   bb   = m0 / T_;              // batch index of this tile
  const int   t0   = m0 % T_;              // first time step
  const int   bblk = bb >> 4, brow = bb & 15;
  const int   n    = lane & 15;
  const float bn   = b_lstm[n0 + n];
#pragma unroll
  for (int v = 0; v < 8; ++v) {
    const int m = (lane < 16) ? v : v + 8;      // local row = time offset
    xk[(((size_t)bblk * T_ + (t0 + m)) * 16 + brow) * G4H + n0 + n] = acc[v] + bn;
  }
}

// ---------------------------------------------------------------------------
// K3: LSTM scan. 16 blocks x 256 threads. Wave w owns hidden cols
// [16w,16w+16): its 4 accumulators are exactly the i/f/g/o gate slices, so
// cell state c never leaves registers. Per-step 32KB gate tile staged into
// LDS by the Tensor Data Mover, overlapped with the WMMA chain of the
// previous step. R_lstm B-fragments preloaded in registers for all 512 steps.
// ---------------------------------------------------------------------------
__global__ void lstm_kernel(const float* __restrict__ xk,
                            const __bf16* __restrict__ R,
                            __bf16* __restrict__ hs) {
  __shared__ float  xkT[16 * 516];         // padded gate tile (33KB)
  __shared__ __bf16 hL[16][H_ + 8];
  const int b0   = blockIdx.x * 16;
  const int wave = wave_id();
  const int lane = threadIdx.x & 31;
  const int n0   = wave * 16;
  const int n    = lane & 15;

  v16bf rb[4][4];
#pragma unroll
  for (int g = 0; g < 4; ++g)
#pragma unroll
    for (int kf = 0; kf < 4; ++kf)
      rb[g][kf] = frag_B_bf16(R, G4H, kf * 32, g * H_ + n0);

  for (int i = threadIdx.x; i < 16 * (H_ + 8); i += blockDim.x)
    (&hL[0][0])[i] = (__bf16)0.0f;
  v8f c = {};

  const size_t tileStride = (size_t)16 * G4H;                // 8192 floats
  const float* tileBase   = xk + (size_t)blockIdx.x * T_ * tileStride;

  stage_xk_tile(tileBase, xkT);            // tile for t = 0
  stage_wait();
  __syncthreads();

  for (int t = 0; t < T_; ++t) {
    v16bf a[4];
#pragma unroll
    for (int kf = 0; kf < 4; ++kf) a[kf] = frag_A_bf16(&hL[0][0], H_ + 8, kf * 32);

    v8f z[4];
#pragma unroll
    for (int g = 0; g < 4; ++g) {
#pragma unroll
      for (int v = 0; v < 8; ++v) {
        const int m = (lane < 16) ? v : v + 8;
        z[g][v] = xkT[xidx(m, g * H_ + n0 + n)];
      }
    }
    __syncthreads();   // all reads of xkT and hL complete

    if (t + 1 < T_)    // DMA next gate tile; overlaps WMMA + gate math below
      stage_xk_tile(tileBase + (size_t)(t + 1) * tileStride, xkT);

#pragma unroll
    for (int g = 0; g < 4; ++g) {
      v8f acc = z[g];
#pragma unroll
      for (int kf = 0; kf < 4; ++kf) acc = wmma_bf16(a[kf], rb[g][kf], acc);
      z[g] = acc;
    }

#pragma unroll
    for (int v = 0; v < 8; ++v) {
      const int   m  = (lane < 16) ? v : v + 8;
      const float iv = sigmoid_(z[0][v]);
      const float fv = sigmoid_(z[1][v]);
      const float gv = tanh_(z[2][v]);
      const float ov = sigmoid_(z[3][v]);
      const float cv = fv * c[v] + iv * gv;
      c[v] = cv;
      const __bf16 hb = (__bf16)(ov * tanh_(cv));
      hL[m][n0 + n] = hb;
      hs[((size_t)(b0 + m) * T_ + t) * H_ + n0 + n] = hb;
    }
    stage_wait();      // next tile landed (wave0's TENSORcnt / ASYNCcnt)
    __syncthreads();   // publish hL writes + staged tile to all waves
  }
}

// ---------------------------------------------------------------------------
// K4: fused recognition heads + generator + KL + reconstruction.
// One block per 16 rows of the flattened (b,t) dimension; writes one partial.
// ---------------------------------------------------------------------------
__global__ void head_kernel(const __bf16* __restrict__ hs,
                            const __bf16* __restrict__ Wrm, const float* __restrict__ brm,
                            const __bf16* __restrict__ Wrs, const float* __restrict__ brs,
                            const __bf16* __restrict__ Wgm, const float* __restrict__ bgm,
                            const __bf16* __restrict__ Wgs, const float* __restrict__ bgs,
                            const float* __restrict__ obs,
                            const float* __restrict__ prior_mean,
                            const float* __restrict__ prior_scale,
                            float* __restrict__ partial) {
  __shared__ __bf16 hT[16][H_ + 8];
  __shared__ float  pmF[16][S_ + 4];
  __shared__ float  psF[16][S_ + 4];
  __shared__ __bf16 pmB[16][S_ + 8];
  __shared__ float  red[256];
  const int m0   = blockIdx.x * 16;
  const int wave = wave_id();
  const int lane = threadIdx.x & 31;
  const int n    = lane & 15;

  // stage 16x128 bf16 hs tile via async-to-LDS (256 x 16B chunks)
  for (int ch = threadIdx.x; ch < 256; ch += blockDim.x) {
    const int r = ch >> 4, w = ch & 15;
    cp16_async(&hT[r][w * 8], hs + (size_t)(m0 + r) * H_ + w * 8);
  }
  wait_async0();
  __syncthreads();

  // phase 1: post_mean / post_scale heads (waves 0-3: mean, 4-7: scale)
  {
    const int mat = wave >> 2;           // scalar
    const int nn0 = (wave & 3) * 16;
    const __bf16* W    = mat ? Wrs : Wrm;
    const float*  bias = mat ? brs : brm;
    v8f acc = {};
#pragma unroll
    for (int kf = 0; kf < 4; ++kf) {
      v16bf a = frag_A_bf16(&hT[0][0], H_ + 8, kf * 32);
      v16bf b = frag_B_bf16(W, S_, kf * 32, nn0);
      acc = wmma_bf16(a, b, acc);
    }
    const float bn = bias[nn0 + n];
#pragma unroll
    for (int v = 0; v < 8; ++v) {
      const int m = (lane < 16) ? v : v + 8;
      const float val = acc[v] + bn;
      if (mat == 0) { pmF[m][nn0 + n] = val; pmB[m][nn0 + n] = (__bf16)val; }
      else          { psF[m][nn0 + n] = softplus_(val); }
    }
  }
  __syncthreads();

  float lsum = 0.0f;   // accumulates kl/(B*T) + (-logp)/B contributions

  // phase 2: generator + reconstruction (wave w -> obs cols [16w,16w+16))
  {
    const int nn0 = wave * 16;
    v8f am = {}, as = {};
#pragma unroll
    for (int kf = 0; kf < 2; ++kf) {
      v16bf a = frag_A_bf16(&pmB[0][0], S_ + 8, kf * 32);
      am = wmma_bf16(a, frag_B_bf16(Wgm, O_, kf * 32, nn0), am);
      as = wmma_bf16(a, frag_B_bf16(Wgs, O_, kf * 32, nn0), as);
    }
    const float bm = bgm[nn0 + n];
    const float bs = bgs[nn0 + n];
#pragma unroll
    for (int v = 0; v < 8; ++v) {
      const int   m  = (lane < 16) ? v : v + 8;
      const float om = am[v] + bm;
      const float os = softplus_(as[v] + bs);
      const float x  = obs[(size_t)(m0 + m) * O_ + nn0 + n];
      const float d  = (x - om) / os;
      const float logp = -0.5f * d * d - flog_(os) - 0.5f * LOG2PI_;
      lsum += (-logp) * (1.0f / (float)B_);
    }
  }

  // phase 3: KL(posterior || prior), all 256 threads over 16x64 elements
  for (int i = threadIdx.x; i < 16 * S_; i += blockDim.x) {
    const int m = i >> 6, cc = i & 63;
    const float mp = pmF[m][cc];
    const float sp = psF[m][cc];
    const float mq = prior_mean[(size_t)(m0 + m) * S_ + cc];
    const float sq = prior_scale[(size_t)(m0 + m) * S_ + cc];
    const float dd = mp - mq;
    const float kl = flog_(sq) - flog_(sp)
                   + (sp * sp + dd * dd) * 0.5f * __builtin_amdgcn_rcpf(sq * sq)
                   - 0.5f;
    lsum += kl * (1.0f / (float)(B_ * T_));
  }

  red[threadIdx.x] = lsum;
  __syncthreads();
  for (int s = 128; s > 0; s >>= 1) {
    if (threadIdx.x < (unsigned)s) red[threadIdx.x] += red[threadIdx.x + s];
    __syncthreads();
  }
  if (threadIdx.x == 0) partial[blockIdx.x] = red[0];
}

// ---------------------------------------------------------------------------
// K5: final deterministic reduction of 8192 block partials
// ---------------------------------------------------------------------------
__global__ void reduce_kernel(const float* __restrict__ partial,
                              float* __restrict__ out, int nPart) {
  __shared__ float red[256];
  float s = 0.0f;
  for (int i = threadIdx.x; i < nPart; i += blockDim.x) s += partial[i];
  red[threadIdx.x] = s;
  __syncthreads();
  for (int k = 128; k > 0; k >>= 1) {
    if (threadIdx.x < (unsigned)k) red[threadIdx.x] += red[threadIdx.x + k];
    __syncthreads();
  }
  if (threadIdx.x == 0) out[0] = red[0];
}

// ---------------------------------------------------------------------------
// Host launcher
// ---------------------------------------------------------------------------
extern "C" void kernel_launch(void* const* d_in, const int* in_sizes, int n_in,
                              void* d_out, int out_size, void* d_ws, size_t ws_size,
                              hipStream_t stream) {
  (void)in_sizes; (void)n_in; (void)out_size; (void)ws_size;
  const float* obs       = (const float*)d_in[0];
  const float* init_mean = (const float*)d_in[1];
  // d_in[2] initial_scale is unused by the reference forward math.
  const float* Wf_m = (const float*)d_in[3];
  const float* bf_m = (const float*)d_in[4];
  const float* Wf_s = (const float*)d_in[5];
  const float* bf_s = (const float*)d_in[6];
  const float* Wg_m = (const float*)d_in[7];
  const float* bg_m = (const float*)d_in[8];
  const float* Wg_s = (const float*)d_in[9];
  const float* bg_s = (const float*)d_in[10];
  const float* K_lstm = (const float*)d_in[11];
  const float* R_lstm = (const float*)d_in[12];
  const float* b_lstm = (const float*)d_in[13];
  const float* Wr_m = (const float*)d_in[14];
  const float* br_m = (const float*)d_in[15];
  const float* Wr_s = (const float*)d_in[16];
  const float* br_s = (const float*)d_in[17];

  char* ws = (char*)d_ws;
  // workspace layout (bytes)
  __bf16* wWfm = (__bf16*)(ws + 0);                    //  8 KB
  __bf16* wWfs = (__bf16*)(ws + 8192);                 //  8 KB
  __bf16* wK   = (__bf16*)(ws + 16384);                // 128 KB
  __bf16* wR   = (__bf16*)(ws + 147456);               // 128 KB
  __bf16* wWrm = (__bf16*)(ws + 278528);               // 16 KB
  __bf16* wWrs = (__bf16*)(ws + 294912);               // 16 KB
  __bf16* wWgm = (__bf16*)(ws + 311296);               // 16 KB
  __bf16* wWgs = (__bf16*)(ws + 327680);               // 16 KB
  float*  pMean  = (float*)(ws + 344064);              // 32 MB
  float*  pScale = (float*)(ws + 33898496ull);         // 32 MB
  float*  xk     = (float*)(ws + 67452928ull);         // 256 MB (tile layout)
  __bf16* hsB    = (__bf16*)(ws + 335888384ull);       // 32 MB
  float*  part   = (float*)(ws + 369442816ull);        // 32 KB

  // 1) weight conversion fp32 -> bf16
  cvt_kernel<<<16, 256, 0, stream>>>(Wf_m, wWfm, S_ * S_);
  cvt_kernel<<<16, 256, 0, stream>>>(Wf_s, wWfs, S_ * S_);
  cvt_kernel<<<256, 256, 0, stream>>>(K_lstm, wK, O_ * G4H);
  cvt_kernel<<<256, 256, 0, stream>>>(R_lstm, wR, H_ * G4H);
  cvt_kernel<<<32, 256, 0, stream>>>(Wr_m, wWrm, H_ * S_);
  cvt_kernel<<<32, 256, 0, stream>>>(Wr_s, wWrs, H_ * S_);
  cvt_kernel<<<32, 256, 0, stream>>>(Wg_m, wWgm, S_ * O_);
  cvt_kernel<<<32, 256, 0, stream>>>(Wg_s, wWgs, S_ * O_);

  // 2) prior rollout (sequential in T, parallel over batch)
  prior_kernel<<<B_ / 16, 256, 0, stream>>>(init_mean, bf_m, bf_s, wWfm, wWfs,
                                            pMean, pScale);

  // 3) hoisted input GEMM: xk = obs @ K_lstm + b_lstm (LSTM tile layout)
  xk_kernel<<<dim3(M_ / 16, G4H / 64), 128, 0, stream>>>(obs, wK, b_lstm, xk);

  // 4) LSTM scan (sequential in T, parallel over batch, TDM-staged gates)
  lstm_kernel<<<B_ / 16, 256, 0, stream>>>(xk, wR, hsB);

  // 5) fused heads + generator + KL + reconstruction -> per-block partials
  head_kernel<<<M_ / 16, 256, 0, stream>>>(hsB, wWrm, br_m, wWrs, br_s,
                                           wWgm, bg_m, wWgs, bg_s,
                                           obs, pMean, pScale, part);

  // 6) final reduction to the scalar loss
  reduce_kernel<<<1, 256, 0, stream>>>(part, (float*)d_out, M_ / 16);
}
